// labelPrediction_cross2_88837103550554
// MI455X (gfx1250) — matrix operand order repaired and checked by reading, not compiled, not run on hardware
//
#include <hip/hip_runtime.h>
#include <hip/hip_bf16.h>

// ---------------------------------------------------------------------------
// CDNA5 (gfx1250, wave32) bf16-WMMA implementation of the labelPrediction
// forward pass.  All GEMMs run as 128x128x32 tiles on v_wmma_f32_16x16x32_bf16
// with f32 accumulation, double-buffered LDS, and async global->LDS staging
// for the A tile (ASYNCcnt / s_wait_asynccnt).  All operands are
// pre-converted to bf16 in workspace with K/N padded to tile multiples, so
// the GEMM inner loop is branch-free.  bias/relu/sigmoid/(scores-thresholds)
// are fused into the GEMM epilogue so the two [4096,5000] tensors are written
// exactly once.
// ---------------------------------------------------------------------------

#define USE_ASYNC_LDS 1

typedef __attribute__((ext_vector_type(16))) __bf16 v16bf;
typedef __attribute__((ext_vector_type(8)))  float  v8f;

#define BM 128
#define BN 128
#define BK 32
#define PADH 8  // bf16 elements of padding per LDS row (16 bytes)

__device__ __forceinline__ unsigned short f32_to_bf16_rte(float f) {
  unsigned int u = __float_as_uint(f);
  unsigned int r = u + 0x7FFFu + ((u >> 16) & 1u);
  return (unsigned short)(r >> 16);
}

// --------------- f32 [R,C] -> bf16 [Rp,Cp] convert with zero pad -----------
__global__ void cvt_pad_bf16_kernel(const float* __restrict__ src,
                                    unsigned short* __restrict__ dst,
                                    int R, int C, int Cp, long totalQuads) {
  long q = (long)blockIdx.x * blockDim.x + threadIdx.x;
  long stride = (long)gridDim.x * blockDim.x;
  for (; q < totalQuads; q += stride) {
    long e = q * 4;
    int r = (int)(e / Cp);
    int c = (int)(e % Cp);
    ushort4 h;
    if (r < R && c + 3 < C) {
      float4 f = *(const float4*)(src + (size_t)r * C + c);
      h.x = f32_to_bf16_rte(f.x);
      h.y = f32_to_bf16_rte(f.y);
      h.z = f32_to_bf16_rte(f.z);
      h.w = f32_to_bf16_rte(f.w);
    } else {
      unsigned short t[4];
      for (int j = 0; j < 4; ++j)
        t[j] = (r < R && c + j < C) ? f32_to_bf16_rte(src[(size_t)r * C + c + j])
                                    : (unsigned short)0;
      h = make_ushort4(t[0], t[1], t[2], t[3]);
    }
    *(ushort4*)(dst + (size_t)r * Cp + c) = h;
  }
}

// ------------------------- zero fill (bf16 buffer) -------------------------
__global__ void zero_u16_kernel(unsigned short* __restrict__ dst, long n) {
  long i = ((long)blockIdx.x * blockDim.x + threadIdx.x) * 8;
  long stride = (long)gridDim.x * blockDim.x * 8;
  uint4 z = make_uint4(0u, 0u, 0u, 0u);
  for (; i + 7 < n; i += stride) *(uint4*)(dst + i) = z;
}

// ------------------- negative label sampling (1..4 per row) ----------------
__global__ void neg_label_kernel(const float* __restrict__ label,
                                 unsigned short* __restrict__ neg,
                                 int C, int Cp, int B) {
  int row = blockIdx.x * blockDim.x + threadIdx.x;
  if (row >= B) return;
  unsigned s = 0x9E3779B9u * (unsigned)(row + 1) ^ 0x85EBCA6Bu;
  auto rnd = [&]() {
    s ^= s << 13; s ^= s >> 17; s ^= s << 5;
    return s;
  };
  int cnt = 1 + (int)(rnd() & 3u);  // randCount in [1,4]
  int picks[4];
  int chosen = 0, attempts = 0;
  while (chosen < cnt && attempts < 64) {
    ++attempts;
    int c = (int)(rnd() % (unsigned)C);
    if (label[(size_t)row * C + c] != 0.0f) continue;  // only zero slots
    bool dup = false;
    for (int i = 0; i < chosen; ++i) dup |= (picks[i] == c);
    if (dup) continue;
    picks[chosen++] = c;
    neg[(size_t)row * Cp + c] = 0x3F80;  // bf16 1.0
  }
}

// ---------------------------------------------------------------------------
// Tiled bf16 GEMM:  C[M,N] = epi(A[M,Kp] @ B[Kp,ldb] + bias[N])
// A is [M,Kp] (Kp % 32 == 0), B is [Kp, ldb] with ldb >= ceil128(N); both
// zero-padded so the K-loop is branch-free.  Double-buffered LDS; A tile
// staged with global_load_async_to_lds_b128, B tile via 4x4 register
// transpose.  Epilogue guards only n < N.
// EPI: 0 = linear, 1 = relu, 2 = sigmoid, 3 = Caux - sigmoid(.)
// Cb (bf16, stride N) and Cf (f32, stride N) outputs are each optional.
// ---------------------------------------------------------------------------
template <int EPI>
__global__ __launch_bounds__(256) void gemm_bf16_wmma_kernel(
    const unsigned short* __restrict__ A,   // [M,Kp] bf16, row major
    const unsigned short* __restrict__ B,   // [Kp,ldb] bf16, row major
    const float* __restrict__ bias,         // [N] f32
    unsigned short* __restrict__ Cb,        // [M,N] bf16 (nullable)
    float* __restrict__ Cf,                 // [M,N] f32  (nullable)
    const float* __restrict__ Caux,         // [M,N] f32  (EPI==3)
    int M, int N, int Kp, int ldb) {
  __shared__ unsigned short As[2][BM][BK + PADH];  // A-tiles, [m][k]
  __shared__ unsigned short Bs[2][BN][BK + PADH];  // B-tiles transposed, [n][k]

  const int tid  = threadIdx.x;
  const int lane = tid & 31;
  const int wave = tid >> 5;   // 8 waves
  const int wr   = wave >> 2;  // 0..1  -> 64-row strip
  const int wc   = wave & 3;   // 0..3  -> 32-col strip
  const int l16  = lane & 15;
  const int lhi  = lane >> 4;  // 0 or 1

  const int mBlk = blockIdx.y * BM;
  const int nBlk = blockIdx.x * BN;

  v8f acc[4][2];
  v8f zero = {0.f, 0.f, 0.f, 0.f, 0.f, 0.f, 0.f, 0.f};
#pragma unroll
  for (int i = 0; i < 4; ++i)
#pragma unroll
    for (int j = 0; j < 2; ++j) acc[i][j] = zero;

  // A staging: 256 threads x 32 bytes = 128x32 tile
  const int aRow = tid >> 1;        // 0..127
  const int aCol = (tid & 1) * 16;  // 0 or 16
  // B staging: 4x4 in-register transpose (4x b64 load, 4x b64 DS store)
  const int bK4 = (tid >> 5) * 4;   // 0,4,...,28
  const int bN4 = (tid & 31) * 4;   // 0..124

  const unsigned short* aPtr = A + (size_t)(mBlk + aRow) * Kp + aCol;
  const unsigned short* bPtr = B + (size_t)bK4 * ldb + (nBlk + bN4);

  auto stage = [&](int buf, int k0) {
#if USE_ASYNC_LDS
    // CDNA5 async copy: bypass VGPRs, track on ASYNCcnt.  INST_OFFSET is
    // applied to both the LDS and the global address, so one address pair
    // covers both 16-byte chunks.
    unsigned ldsAddr = (unsigned)(size_t)&As[buf][aRow][aCol];
    unsigned long long ga = (unsigned long long)(size_t)(aPtr + k0);
    asm volatile(
        "global_load_async_to_lds_b128 %0, %1, off\n\t"
        "global_load_async_to_lds_b128 %0, %1, off offset:16" ::"v"(ldsAddr),
        "v"(ga)
        : "memory");
#else
    const uint4* asrc = (const uint4*)(aPtr + k0);
    uint4 a0 = asrc[0];
    uint4 a1 = asrc[1];
    *(uint4*)&As[buf][aRow][aCol]     = a0;
    *(uint4*)&As[buf][aRow][aCol + 8] = a1;
#endif
    const unsigned short* bsrc = bPtr + (size_t)k0 * ldb;
    ushort4 r0 = *(const ushort4*)(bsrc);
    ushort4 r1 = *(const ushort4*)(bsrc + ldb);
    ushort4 r2 = *(const ushort4*)(bsrc + 2 * ldb);
    ushort4 r3 = *(const ushort4*)(bsrc + 3 * ldb);
    *(ushort4*)&Bs[buf][bN4 + 0][bK4] = make_ushort4(r0.x, r1.x, r2.x, r3.x);
    *(ushort4*)&Bs[buf][bN4 + 1][bK4] = make_ushort4(r0.y, r1.y, r2.y, r3.y);
    *(ushort4*)&Bs[buf][bN4 + 2][bK4] = make_ushort4(r0.z, r1.z, r2.z, r3.z);
    *(ushort4*)&Bs[buf][bN4 + 3][bK4] = make_ushort4(r0.w, r1.w, r2.w, r3.w);
  };

  // prologue: stage buffer 0
  stage(0, 0);
#if USE_ASYNC_LDS
  asm volatile("s_wait_asynccnt 0x0" ::: "memory");
#endif
  __syncthreads();

  for (int k0 = 0; k0 < Kp; k0 += BK) {
    const int cur = (k0 >> 5) & 1;
    const int nxt = cur ^ 1;
    const bool more = (k0 + BK) < Kp;
    if (more) {
      stage(nxt, k0 + BK);  // overlaps with WMMA below
      if (k0 + 2 * BK < Kp)
        __builtin_prefetch(bPtr + (size_t)(k0 + 2 * BK) * ldb, 0, 1);
    }

    // ---- load fragments from current buffer (A and B both in contiguous k order, k = 16*lhi + s) ----
    union Frag { v16bf v; uint4 q[2]; };
    Frag a[4], b[2];
#pragma unroll
    for (int mt = 0; mt < 4; ++mt) {
      int r  = wr * 64 + mt * 16 + l16;  // A: both lane halves cover M=0..15
      int ko = lhi * 16;                 // lanes 0-15: K 0..15, 16-31: K 16..31
      a[mt].q[0] = *(const uint4*)&As[cur][r][ko];
      a[mt].q[1] = *(const uint4*)&As[cur][r][ko + 8];
    }
#pragma unroll
    for (int nt = 0; nt < 2; ++nt) {
      int c  = wc * 32 + nt * 16 + l16;  // B: lane = column
      int ko = lhi * 16;                 // lanes 0-15: K 0..15, 16-31: K 16..31
      b[nt].q[0] = *(const uint4*)&Bs[cur][c][ko];
      b[nt].q[1] = *(const uint4*)&Bs[cur][c][ko + 8];
    }

    // ---- 8x WMMA per K-step ----
#pragma unroll
    for (int mt = 0; mt < 4; ++mt)
#pragma unroll
      for (int nt = 0; nt < 2; ++nt)
        acc[mt][nt] = __builtin_amdgcn_wmma_f32_16x16x32_bf16(
            false, a[mt].v, false, b[nt].v, (short)0, acc[mt][nt], false,
            false);

#if USE_ASYNC_LDS
    if (more) asm volatile("s_wait_asynccnt 0x0" ::: "memory");
#endif
    __syncthreads();
  }

  // ---- fused epilogue; C/D layout: lanes 0-15 -> M=g, lanes 16-31 -> M=8+g
#pragma unroll
  for (int nt = 0; nt < 2; ++nt) {
    int n = nBlk + wc * 32 + nt * 16 + l16;
    bool nOK = (n < N);
    float bv = 0.f;
    if (bias != nullptr && nOK) bv = bias[n];
#pragma unroll
    for (int mt = 0; mt < 4; ++mt) {
      int mBase = mBlk + wr * 64 + mt * 16 + lhi * 8;
      v8f v = acc[mt][nt];
#pragma unroll
      for (int g = 0; g < 8; ++g) {
        if (nOK) {
          int m = mBase + g;
          float x = v[g] + bv;
          size_t idx = (size_t)m * N + n;
          if (EPI == 1) x = x > 0.f ? x : 0.f;
          if (EPI == 2) x = 1.f / (1.f + __expf(-x));
          if (EPI == 3) x = Caux[idx] - (1.f / (1.f + __expf(-x)));
          if (Cb) Cb[idx] = f32_to_bf16_rte(x);
          if (Cf) Cf[idx] = x;
        }
      }
    }
  }
}

// ---------------------------------------------------------------------------
extern "C" void kernel_launch(void* const* d_in, const int* in_sizes, int n_in,
                              void* d_out, int out_size, void* d_ws,
                              size_t ws_size, hipStream_t stream) {
  (void)in_sizes; (void)n_in; (void)out_size; (void)ws_size;

  const float* visual = (const float*)d_in[0];
  const float* label  = (const float*)d_in[1];
  const float* ve_w1 = (const float*)d_in[2];
  const float* ve_b1 = (const float*)d_in[3];
  const float* ve_w2 = (const float*)d_in[4];
  const float* ve_b2 = (const float*)d_in[5];
  const float* ve_w3 = (const float*)d_in[6];
  const float* ve_b3 = (const float*)d_in[7];
  const float* le_w1 = (const float*)d_in[8];
  const float* le_b1 = (const float*)d_in[9];
  const float* le_w2 = (const float*)d_in[10];
  const float* le_b2 = (const float*)d_in[11];
  const float* cl_w1 = (const float*)d_in[12];
  const float* cl_b1 = (const float*)d_in[13];
  const float* cl_w2 = (const float*)d_in[14];
  const float* cl_b2 = (const float*)d_in[15];
  const float* th_w1 = (const float*)d_in[16];
  const float* th_b1 = (const float*)d_in[17];
  const float* th_w2 = (const float*)d_in[18];
  const float* th_b2 = (const float*)d_in[19];

  const int Bn = 4096, CLS = 5000;
  const int CLS_K = 5024;   // label K padded to mult of 32
  const int CLS_N = 5120;   // class N padded to mult of 128

  // ---- workspace bump allocator (bf16 buffers) ----
  size_t off = 0;
  auto alloc = [&](size_t elems) -> unsigned short* {
    unsigned short* p = (unsigned short*)((char*)d_ws + off);
    off += ((elems * sizeof(unsigned short)) + 255) & ~(size_t)255;
    return p;
  };
  unsigned short* visb = alloc((size_t)Bn * 2048);
  unsigned short* labb = alloc((size_t)Bn * CLS_K);     // [4096,5024]
  unsigned short* negb = alloc((size_t)Bn * CLS_K);     // [4096,5024]
  unsigned short* w1b  = alloc((size_t)2048 * 1024);
  unsigned short* w2b  = alloc((size_t)1024 * 512);
  unsigned short* w3b  = alloc((size_t)512 * 256);
  unsigned short* lw1b = alloc((size_t)CLS_K * 128);    // [5024,128]
  unsigned short* lw2b = alloc((size_t)128 * 256);
  unsigned short* cw1b = alloc((size_t)256 * 128);
  unsigned short* cw2b = alloc((size_t)128 * CLS_N);    // [128,5120]
  unsigned short* tw1b = alloc((size_t)256 * 128);
  unsigned short* tw2b = alloc((size_t)128 * CLS_N);    // [128,5120]
  unsigned short* h1b  = alloc((size_t)Bn * 1024);
  unsigned short* h2b  = alloc((size_t)Bn * 512);
  unsigned short* evb  = alloc((size_t)Bn * 256);
  unsigned short* p1b  = alloc((size_t)Bn * 128);
  unsigned short* n1b  = alloc((size_t)Bn * 128);
  unsigned short* c1b  = alloc((size_t)Bn * 128);
  unsigned short* t1b  = alloc((size_t)Bn * 128);

  // ---- output layout (flat f32 concat in reference return order) ----
  float* out        = (float*)d_out;
  float* out_scores = out;
  float* out_margin = out_scores + (size_t)Bn * CLS;
  float* out_ev     = out_margin + (size_t)Bn * CLS;
  float* out_pos    = out_ev + (size_t)Bn * 256;
  float* out_neg    = out_pos + (size_t)Bn * 256;

  auto cvt = [&](const float* s, unsigned short* d, int R, int C, int Rp,
                 int Cp) {
    long quads = (long)Rp * Cp / 4;
    long blocks = (quads + 255) / 256;
    if (blocks < 1) blocks = 1;
    if (blocks > 16384) blocks = 16384;
    cvt_pad_bf16_kernel<<<dim3((unsigned)blocks), dim3(256), 0, stream>>>(
        s, d, R, C, Cp, quads);
  };

  // ---- bf16 conversions (with zero padding where needed) ----
  cvt(visual, visb, Bn, 2048, Bn, 2048);
  cvt(label, labb, Bn, CLS, Bn, CLS_K);
  cvt(ve_w1, w1b, 2048, 1024, 2048, 1024);
  cvt(ve_w2, w2b, 1024, 512, 1024, 512);
  cvt(ve_w3, w3b, 512, 256, 512, 256);
  cvt(le_w1, lw1b, CLS, 128, CLS_K, 128);
  cvt(le_w2, lw2b, 128, 256, 128, 256);
  cvt(cl_w1, cw1b, 256, 128, 256, 128);
  cvt(cl_w2, cw2b, 128, CLS, 128, CLS_N);
  cvt(th_w1, tw1b, 256, 128, 256, 128);
  cvt(th_w2, tw2b, 128, CLS, 128, CLS_N);

  // ---- negative labels: zero (padded) then scatter 1..4 ones per row ----
  {
    long n = (long)Bn * CLS_K;
    long blocks = (n / 8 + 255) / 256;
    if (blocks > 16384) blocks = 16384;
    zero_u16_kernel<<<dim3((unsigned)blocks), dim3(256), 0, stream>>>(negb, n);
    neg_label_kernel<<<dim3(16), dim3(256), 0, stream>>>(label, negb, CLS,
                                                         CLS_K, Bn);
  }

  auto gemm = [&](int epi, const unsigned short* A, const unsigned short* W,
                  const float* bias, unsigned short* Cb, float* Cf,
                  const float* Caux, int M, int N, int Kp, int ldb) {
    dim3 g((N + BN - 1) / BN, (M + BM - 1) / BM), b(256);
    switch (epi) {
      case 0: gemm_bf16_wmma_kernel<0><<<g, b, 0, stream>>>(A, W, bias, Cb, Cf, Caux, M, N, Kp, ldb); break;
      case 1: gemm_bf16_wmma_kernel<1><<<g, b, 0, stream>>>(A, W, bias, Cb, Cf, Caux, M, N, Kp, ldb); break;
      case 2: gemm_bf16_wmma_kernel<2><<<g, b, 0, stream>>>(A, W, bias, Cb, Cf, Caux, M, N, Kp, ldb); break;
      case 3: gemm_bf16_wmma_kernel<3><<<g, b, 0, stream>>>(A, W, bias, Cb, Cf, Caux, M, N, Kp, ldb); break;
    }
  };

  // ---- visual encoder: 2048 -> 1024 -> 512 -> 256 ----
  gemm(1, visb, w1b, ve_b1, h1b, nullptr, nullptr, Bn, 1024, 2048, 1024);
  gemm(1, h1b, w2b, ve_b2, h2b, nullptr, nullptr, Bn, 512, 1024, 512);
  gemm(0, h2b, w3b, ve_b3, evb, out_ev, nullptr, Bn, 256, 512, 256);

  // ---- label encoder (positive): 5000 -> 128 -> 256 ----
  gemm(1, labb, lw1b, le_b1, p1b, nullptr, nullptr, Bn, 128, CLS_K, 128);
  gemm(0, p1b, lw2b, le_b2, nullptr, out_pos, nullptr, Bn, 256, 128, 256);

  // ---- label encoder (negative) ----
  gemm(1, negb, lw1b, le_b1, n1b, nullptr, nullptr, Bn, 128, CLS_K, 128);
  gemm(0, n1b, lw2b, le_b2, nullptr, out_neg, nullptr, Bn, 256, 128, 256);

  // ---- classifier: scores = sigmoid(relu(ev@cl_w1+b)@cl_w2+b) ----
  gemm(1, evb, cw1b, cl_b1, c1b, nullptr, nullptr, Bn, 128, 256, 128);
  gemm(2, c1b, cw2b, cl_b2, nullptr, out_scores, nullptr, Bn, CLS, 128, CLS_N);

  // ---- thresholds fused: margin = scores - sigmoid(relu(ev@th_w1+b)@th_w2+b)
  gemm(1, evb, tw1b, th_b1, t1b, nullptr, nullptr, Bn, 128, 256, 128);
  gemm(3, t1b, tw2b, th_b2, nullptr, out_margin, out_scores, Bn, CLS, 128, CLS_N);
}
